// BaseContrastiveLoss_1795296330338
// MI455X (gfx1250) — matrix operand order repaired and checked by reading, not compiled, not run on hardware
//
#include <hip/hip_runtime.h>
#include <hip/hip_bf16.h>

typedef __attribute__((ext_vector_type(16))) __bf16 v16bf;
typedef __attribute__((ext_vector_type(8)))  float  v8f;

#define N_ROWS   8192
#define DIMK     1024
#define NTILES   512              // 8192 / 16
#define TEMP_INV (1.0f / 0.07f)

// ---------------------------------------------------------------------------
// Kernel 1: L2-normalize each row (fp32) and emit bf16 into workspace.
// ---------------------------------------------------------------------------
__global__ __launch_bounds__(256) void nrm_kernel(const float* __restrict__ f,
                                                  __bf16* __restrict__ out) {
  const int row = blockIdx.x;
  const int t   = threadIdx.x;
  const float* fr = f + (size_t)row * DIMK;

  float v[4];
  float ss = 0.f;
#pragma unroll
  for (int i = 0; i < 4; ++i) {
    v[i] = fr[t + i * 256];
    ss += v[i] * v[i];
  }
#pragma unroll
  for (int m = 16; m >= 1; m >>= 1) ss += __shfl_xor(ss, m, 32);

  __shared__ float warp_ss[8];
  if ((t & 31) == 0) warp_ss[t >> 5] = ss;
  __syncthreads();

  float tot = 0.f;
#pragma unroll
  for (int w = 0; w < 8; ++w) tot += warp_ss[w];
  const float rn = rsqrtf(tot);

  __bf16* orow = out + (size_t)row * DIMK;
#pragma unroll
  for (int i = 0; i < 4; ++i) orow[t + i * 256] = (__bf16)(v[i] * rn);
}

// ---------------------------------------------------------------------------
// Kernel 2: fused  sim = (fn @ fn^T)/T  + masked row sum-exp + positives.
// Block = 256 threads (8 waves), one block per 32-row M-panel (2 M-tiles).
// A-panel (32 x 1024 bf16 = 64 KB) staged in LDS, reused for all N-tiles.
// Each wave: 64-N-tile strip, processed as 32 pairs with 2x2 register
// blocking -> 4 WMMAs per K-step sharing 2 A- and 2 B-fragments.
// ---------------------------------------------------------------------------
__global__ __launch_bounds__(256) void ntxent_kernel(const __bf16* __restrict__ fn,
                                                     float* __restrict__ out) {
  // 64 KB buffer: A-panel during GEMM, then aliased as reduction scratch.
  __shared__ __align__(16) unsigned char lds_raw[32 * DIMK * 2];
  __bf16* lds_a = (__bf16*)lds_raw;

  const int mtile0   = blockIdx.x * 2;               // even M-tile
  const int postile0 = (mtile0 + 256) & (NTILES - 1);
  const int t    = threadIdx.x;
  const int wave = t >> 5;
  const int lane = t & 31;
  const int lm   = lane & 15;
  const int lh   = lane >> 4;

  // ---- stage 32-row A-panel into LDS (4096 uint4 across 256 threads) ----
  {
    const uint4* src = (const uint4*)(fn + (size_t)mtile0 * 16 * DIMK);
    uint4* dst = (uint4*)lds_raw;
#pragma unroll
    for (int i = 0; i < 16; ++i) dst[t + i * 256] = src[t + i * 256];
  }
  __syncthreads();

  // A fragment bases (16-bit A 16x32 VGPR layout):
  //   elems 0..7  <- K = k0 + lh*8 .. +7 ; elems 8..15 <- K = k0+16+lh*8 .. +7
  const __bf16* a_base0 = lds_a + (0 * 16 + lm) * DIMK + lh * 8;
  const __bf16* a_base1 = lds_a + (1 * 16 + lm) * DIMK + lh * 8;

  float S[2][8];
  float posv[2] = {0.f, 0.f};
#pragma unroll
  for (int mi = 0; mi < 2; ++mi)
#pragma unroll
    for (int r = 0; r < 8; ++r) S[mi][r] = 0.f;

  // Lane owns the tile diagonal element at r == diag_r (lanes 0..7, 24..31).
  const int diag_r = lm - 8 * lh;

  const int nt0 = wave * 64;
  const int nt1 = nt0 + 64;

  for (int nt = nt0; nt < nt1; nt += 2) {
    // B fragments (32x16 bf16): lane = column n, K = lh*8..+7 and 16+lh*8..+7.
    const __bf16* b_base0 = fn + ((size_t)((nt + 0) * 16 + lm)) * DIMK + lh * 8;
    const __bf16* b_base1 = fn + ((size_t)((nt + 1) * 16 + lm)) * DIMK + lh * 8;

    // Prefetch next pair's B rows (lowers to global_prefetch_b8).
    if (nt + 2 < nt1) {
      __builtin_prefetch(fn + ((size_t)((nt + 2) * 16 + lm)) * DIMK + lh * 16, 0, 3);
      __builtin_prefetch(fn + ((size_t)((nt + 3) * 16 + lm)) * DIMK + lh * 16, 0, 3);
    }

    v8f c00 = {0.f,0.f,0.f,0.f,0.f,0.f,0.f,0.f};
    v8f c01 = c00, c10 = c00, c11 = c00;

#pragma unroll 2
    for (int k0 = 0; k0 < DIMK; k0 += 32) {
      union F { uint4 u[2]; v16bf v; } A0, A1, B0, B1;
      A0.u[0] = *(const uint4*)(a_base0 + k0);
      A0.u[1] = *(const uint4*)(a_base0 + k0 + 16);
      A1.u[0] = *(const uint4*)(a_base1 + k0);
      A1.u[1] = *(const uint4*)(a_base1 + k0 + 16);
      const uint4* bp0 = (const uint4*)(b_base0 + k0);
      const uint4* bp1 = (const uint4*)(b_base1 + k0);
      B0.u[0] = bp0[0]; B0.u[1] = bp0[2];
      B1.u[0] = bp1[0]; B1.u[1] = bp1[2];

      c00 = __builtin_amdgcn_wmma_f32_16x16x32_bf16(false, A0.v, false, B0.v,
                                                    (short)0, c00, false, false);
      c01 = __builtin_amdgcn_wmma_f32_16x16x32_bf16(false, A0.v, false, B1.v,
                                                    (short)0, c01, false, false);
      c10 = __builtin_amdgcn_wmma_f32_16x16x32_bf16(false, A1.v, false, B0.v,
                                                    (short)0, c10, false, false);
      c11 = __builtin_amdgcn_wmma_f32_16x16x32_bf16(false, A1.v, false, B1.v,
                                                    (short)0, c11, false, false);
    }

    // ---- epilogue: scale, exp, self-mask, positive capture, accumulate ----
#pragma unroll
    for (int mi = 0; mi < 2; ++mi) {
      const int mt = mtile0 + mi;
      const int pt = postile0 + mi;
#pragma unroll
      for (int nj = 0; nj < 2; ++nj) {
        const int ntj = nt + nj;
        const v8f& c = (mi == 0) ? (nj == 0 ? c00 : c01)
                                 : (nj == 0 ? c10 : c11);
        const bool selfT = (ntj == mt);
        const bool posT  = (ntj == pt);
#pragma unroll
        for (int r = 0; r < 8; ++r) {
          const float v = c[r] * TEMP_INV;
          float e = __expf(v);
          if (selfT && r == diag_r) e = 0.f;
          if (posT  && r == diag_r) posv[mi] = v;
          S[mi][r] += e;
        }
      }
    }
  }

  // All waves done reading the A-panel; safe to alias LDS for reductions.
  __syncthreads();
  float* sums_lds = (float*)lds_raw;          // [8 waves][32 rows]
  float* pos_lds  = sums_lds + 8 * 32;        // [32 rows]

  // Row sums: reduce across the 16 columns of each half-wave group.
#pragma unroll
  for (int mi = 0; mi < 2; ++mi) {
#pragma unroll
    for (int r = 0; r < 8; ++r) {
#pragma unroll
      for (int m = 1; m < 16; m <<= 1) S[mi][r] += __shfl_xor(S[mi][r], m, 32);
    }
  }
  if (lm == 0) {
#pragma unroll
    for (int mi = 0; mi < 2; ++mi)
#pragma unroll
      for (int r = 0; r < 8; ++r)
        sums_lds[wave * 32 + mi * 16 + r + 8 * lh] = S[mi][r];
  }
  // Positive logits: diagonal lanes (row_off == lm) of the owning wave.
  if (diag_r >= 0 && diag_r < 8) {
#pragma unroll
    for (int mi = 0; mi < 2; ++mi) {
      const int pt = postile0 + mi;
      if (pt >= nt0 && pt < nt1) pos_lds[mi * 16 + lm] = posv[mi];
    }
  }
  __syncthreads();

  // ---- final per-row loss for the 32 rows of this block ----
  if (t < 32) {
    float tot = 0.f;
#pragma unroll
    for (int w = 0; w < 8; ++w) tot += sums_lds[w * 32 + t];
    const float loss = __logf(tot) - pos_lds[t];
    atomicAdd(out, loss * (1.0f / (float)N_ROWS));
  }
}

// ---------------------------------------------------------------------------
extern "C" void kernel_launch(void* const* d_in, const int* in_sizes, int n_in,
                              void* d_out, int out_size, void* d_ws, size_t ws_size,
                              hipStream_t stream) {
  const float* feats = (const float*)d_in[0];
  float* out = (float*)d_out;
  __bf16* fnorm = (__bf16*)d_ws;   // 8192*1024*2 = 16 MB scratch

  hipMemsetAsync(d_out, 0, sizeof(float), stream);
  nrm_kernel<<<N_ROWS, 256, 0, stream>>>(feats, fnorm);
  ntxent_kernel<<<NTILES / 2, 256, 0, stream>>>(fnorm, out);
}
